// SlidingWindowAttention_8323646620089
// MI455X (gfx1250) — compile-verified, matrix operand order checked by reading, not run on hardware
//
#include <hip/hip_runtime.h>
#include <hip/hip_bf16.h>
#include <math.h>

// ---------------------------------------------------------------------------
// SlidingWindowAttention for gfx1250 (MI455X): all matmuls via
// v_wmma_f32_16x16x32_f16 (wave32 WMMA, f32 accumulate).
// B=1, T=2048, C=512, NH=8, NKV=4, HD=64, W=128.
// ---------------------------------------------------------------------------

typedef __attribute__((ext_vector_type(16))) _Float16 v16h;
typedef __attribute__((ext_vector_type(8)))  _Float16 v8h;
typedef __attribute__((ext_vector_type(8)))  float    v8f;

union AFrag { v16h v; v8h h[2]; };

#define T_LEN 2048
#define C_DIM 512
#define NH    8
#define NKV   4
#define HD    64
#define WIN   128

// ---------------------------------------------------------------------------
// Kernel 1: convert x -> f16, pack W_q|W_k|W_v transposed into WT[N=1024][K=512],
// and W_o transposed into WoT[N=512][K=512].  [N][K] layout makes WMMA
// B-fragment loads contiguous 128-bit spans per lane.
// ---------------------------------------------------------------------------
__global__ void swa_convert_kernel(const float* __restrict__ x,
                                   const float* __restrict__ Wq,
                                   const float* __restrict__ Wk,
                                   const float* __restrict__ Wv,
                                   const float* __restrict__ Wo,
                                   _Float16* __restrict__ xh,
                                   _Float16* __restrict__ WT,
                                   _Float16* __restrict__ WoT) {
    int idx = blockIdx.x * blockDim.x + threadIdx.x;
    int stride = gridDim.x * blockDim.x;
    for (int i = idx; i < T_LEN * C_DIM; i += stride)
        xh[i] = (_Float16)x[i];
    for (int i = idx; i < 1024 * 512; i += stride) {
        int n = i >> 9, k = i & 511;
        float v;
        if (n < 512)      v = Wq[k * 512 + n];
        else if (n < 768) v = Wk[k * 256 + (n - 512)];
        else              v = Wv[k * 256 + (n - 768)];
        WT[i] = (_Float16)v;
    }
    for (int i = idx; i < 512 * 512; i += stride) {
        int n = i >> 9, k = i & 511;
        WoT[i] = (_Float16)Wo[k * 512 + n];
    }
}

// ---------------------------------------------------------------------------
// Kernel 2: generic f16 WMMA GEMM.  C[M][N] = A[M][K] * Bt[N][K]^T, f32 accum.
// One wave computes a 32x64 tile (2 M-subtiles x 4 N-subtiles = 8 accumulators);
// each B fragment is reused by both A fragments -> 8 WMMA per 12 B128 loads
// per K=32 step.  Fragment layout (wave32, 16-bit A/B, CDNA5 ISA 7.12.2):
//   lanes 0-15 : row/col = lane,     K in {0..7} (h[0]) and {16..23} (h[1])
//   lanes 16-31: row/col = lane-16,  K in {8..15}        and {24..31}
// ---------------------------------------------------------------------------
__global__ __launch_bounds__(256)
void swa_wmma_gemm_f16(const _Float16* __restrict__ A,
                       const _Float16* __restrict__ Bt,
                       float* __restrict__ C,
                       int M, int N, int K, int lda, int ldb, int ldc) {
    int lane = threadIdx.x & 31;
    int wave = (blockIdx.x * blockDim.x + threadIdx.x) >> 5;
    int tilesN = N >> 6;
    int m0 = (wave / tilesN) << 5;   // 32 rows per wave
    int n0 = (wave % tilesN) << 6;   // 64 cols per wave
    if (m0 >= M) return;

    int r    = lane & 15;
    int koff = (lane >> 4) << 3;     // 0 or 8

    const _Float16* pa0 = A + (size_t)(m0 + r) * lda + koff;
    const _Float16* pa1 = pa0 + (size_t)16 * lda;

    v8f acc[2][4] = {};
    for (int k0 = 0; k0 < K; k0 += 32) {
        AFrag a0, a1;
        a0.h[0] = *(const v8h*)(pa0 + k0);
        a0.h[1] = *(const v8h*)(pa0 + k0 + 16);
        a1.h[0] = *(const v8h*)(pa1 + k0);
        a1.h[1] = *(const v8h*)(pa1 + k0 + 16);
        // prefetch next K-chunk of A (lowers to global_prefetch_b8)
        __builtin_prefetch(pa0 + k0 + 32, 0, 1);
        __builtin_prefetch(pa1 + k0 + 32, 0, 1);
#pragma unroll
        for (int s = 0; s < 4; ++s) {
            AFrag b;
            const _Float16* pb = Bt + (size_t)(n0 + (s << 4) + r) * ldb + k0 + koff;
            b.h[0] = *(const v8h*)pb;
            b.h[1] = *(const v8h*)(pb + 16);
            __builtin_prefetch(pb + 32, 0, 1);
            acc[0][s] = __builtin_amdgcn_wmma_f32_16x16x32_f16(
                false, a0.v, false, b.v, (short)0, acc[0][s], false, false);
            acc[1][s] = __builtin_amdgcn_wmma_f32_16x16x32_f16(
                false, a1.v, false, b.v, (short)0, acc[1][s], false, false);
        }
    }
    int col  = lane & 15;
    int moff = (lane >> 4) << 3;
#pragma unroll
    for (int mt = 0; mt < 2; ++mt)
#pragma unroll
        for (int s = 0; s < 4; ++s)
#pragma unroll
            for (int rr = 0; rr < 8; ++rr)
                C[(size_t)(m0 + (mt << 4) + moff + rr) * ldc + n0 + (s << 4) + col] =
                    acc[mt][s][rr];
}

// ---------------------------------------------------------------------------
// Kernel 3: RoPE + RMS-norm on q,k; transpose-copy of v.
// One wave per (t, slot).  slot 0..7: q head; 8..11: k head; 12..15: v head.
// Lane i owns the rotation pair (i, i+32); cos/sin[t][i] maps 1:1 to lanes.
// Outputs: qh[NH][T][HD] f16, kh[NKV][T][HD] f16, vT[NKV][HD][T] f16.
// ---------------------------------------------------------------------------
__global__ __launch_bounds__(256)
void swa_rope_rms_kernel(const float* __restrict__ qkv,
                         const float* __restrict__ cosb,
                         const float* __restrict__ sinb,
                         const float* __restrict__ qw,
                         const float* __restrict__ kw,
                         _Float16* __restrict__ qh,
                         _Float16* __restrict__ kh,
                         _Float16* __restrict__ vT) {
    int lane = threadIdx.x & 31;
    int wave = (blockIdx.x * blockDim.x + threadIdx.x) >> 5;
    int t = wave >> 4;
    int s = wave & 15;
    if (t >= T_LEN) return;

    if (s < 12) {
        int col0;
        const float* wvec;
        _Float16* dst;
        if (s < 8) { col0 = s * 64;              wvec = qw; dst = qh + ((size_t)s * T_LEN + t) * HD; }
        else       { col0 = 512 + (s - 8) * 64;  wvec = kw; dst = kh + ((size_t)(s - 8) * T_LEN + t) * HD; }
        float x1 = qkv[(size_t)t * 1024 + col0 + lane];
        float x2 = qkv[(size_t)t * 1024 + col0 + 32 + lane];
        float c  = cosb[t * 32 + lane];
        float sn = sinb[t * 32 + lane];
        float o1 = x1 * c - x2 * sn;
        float o2 = x2 * c + x1 * sn;
        float ss = o1 * o1 + o2 * o2;
#pragma unroll
        for (int off = 16; off > 0; off >>= 1)
            ss += __shfl_xor(ss, off, 32);
        float rr = rsqrtf(ss * (1.0f / 64.0f) + 1e-6f);
        dst[lane]      = (_Float16)(o1 * rr * wvec[lane]);
        dst[lane + 32] = (_Float16)(o2 * rr * wvec[lane + 32]);
    } else {
        int head = s - 12;
        int col0 = 768 + head * 64;
        float v1 = qkv[(size_t)t * 1024 + col0 + lane];
        float v2 = qkv[(size_t)t * 1024 + col0 + 32 + lane];
        vT[((size_t)head * HD + lane) * T_LEN + t]        = (_Float16)v1;
        vT[((size_t)head * HD + lane + 32) * T_LEN + t]   = (_Float16)v2;
    }
}

// ---------------------------------------------------------------------------
// Kernel 4: sliding-window attention core.  One wave per (head, 16-query tile).
// 9 key tiles of 16 cover the aligned window [t0-128, t0+15]; masking in
// softmax handles key<0 / out-of-window.  P padded to 160 cols (5 K-chunks).
// ---------------------------------------------------------------------------
#define ATTN_WAVES 2
__global__ __launch_bounds__(ATTN_WAVES * 32)
void swa_attn_kernel(const _Float16* __restrict__ qh,
                     const _Float16* __restrict__ kh,
                     const _Float16* __restrict__ vT,
                     _Float16* __restrict__ oh) {
    __shared__ __align__(16) float    sS[ATTN_WAVES][16][160];
    __shared__ __align__(16) _Float16 sP[ATTN_WAVES][16][160];

    int lane  = threadIdx.x & 31;
    int wslot = threadIdx.x >> 5;
    int wave  = blockIdx.x * ATTN_WAVES + wslot;
    int h  = wave >> 7;          // / (T/16 = 128)
    int qt = wave & 127;
    int t0 = qt << 4;
    int kv = h >> 1;             // GQA group of 2

    int r    = lane & 15;
    int koff = (lane >> 4) << 3;
    int moff = (lane >> 4) << 3;

    // --- Q fragments (16 rows x 64 dims -> two K=32 A-frags) ---
    AFrag aq0, aq1;
    {
        const _Float16* pq = qh + ((size_t)h * T_LEN + t0 + r) * HD + koff;
        aq0.h[0] = *(const v8h*)pq;
        aq0.h[1] = *(const v8h*)(pq + 16);
        aq1.h[0] = *(const v8h*)(pq + 32);
        aq1.h[1] = *(const v8h*)(pq + 48);
    }

    int kbase = t0 - WIN;        // aligned window start (may be negative)
    const _Float16* kheadp = kh + (size_t)kv * T_LEN * HD;

    // --- S = Q * K^T over 9 key tiles ---
    for (int j = 0; j < 9; ++j) {
        int key = kbase + (j << 4) + r;   // lane's key row (B-frag N = key)
        AFrag b0, b1;
        if (key >= 0) {
            const _Float16* pk = kheadp + (size_t)key * HD + koff;
            b0.h[0] = *(const v8h*)pk;
            b0.h[1] = *(const v8h*)(pk + 16);
            b1.h[0] = *(const v8h*)(pk + 32);
            b1.h[1] = *(const v8h*)(pk + 48);
        } else {
            v8h z = {};
            b0.h[0] = z; b0.h[1] = z; b1.h[0] = z; b1.h[1] = z;
        }
        v8f acc = {};
        acc = __builtin_amdgcn_wmma_f32_16x16x32_f16(
            false, aq0.v, false, b0.v, (short)0, acc, false, false);
        acc = __builtin_amdgcn_wmma_f32_16x16x32_f16(
            false, aq1.v, false, b1.v, (short)0, acc, false, false);
#pragma unroll
        for (int rr = 0; rr < 8; ++rr)
            sS[wslot][moff + rr][(j << 4) + r] = acc[rr] * 0.125f; // 1/sqrt(64)
    }

    // --- masked softmax: lanes 0..15 each own one query row ---
    if (lane < 16) {
        int i = lane;
        int jlo = i + 1;                       // key >= t-127  ->  j >= i+1
        int lim = WIN - t0;                    // key >= 0      ->  j >= 128-t0
        if (lim > jlo) jlo = lim;
        int jhi = i + WIN;                     // key <= t      ->  j <= i+128
        float m = -3.4e38f;
        for (int j = jlo; j <= jhi; ++j) {
            float v = sS[wslot][i][j];
            if (v > m) m = v;
        }
        float sum = 0.0f;
        for (int j = jlo; j <= jhi; ++j) {
            float e = __expf(sS[wslot][i][j] - m);
            sS[wslot][i][j] = e;
            sum += e;
        }
        float inv = 1.0f / sum;
        for (int j = 0; j < 160; ++j) {
            float p = (j >= jlo && j <= jhi) ? sS[wslot][i][j] * inv : 0.0f;
            sP[wslot][i][j] = (_Float16)p;
        }
    }

    // --- O = P * V : 5 K-chunks of 32 keys x 4 dim tiles of 16 ---
    v8f o[4] = {};
    for (int c = 0; c < 5; ++c) {
        AFrag ap;
        const _Float16* pp = &sP[wslot][r][(c << 5) + koff];
        ap.h[0] = *(const v8h*)pp;
        ap.h[1] = *(const v8h*)(pp + 16);
        int kstart = kbase + (c << 5) + koff;      // 8-aligned span start
#pragma unroll
        for (int s = 0; s < 4; ++s) {
            int dim = kv * HD + (s << 4) + r;      // B-frag N = dim
            AFrag bv;
            v8h z = {};
            const _Float16* pv = vT + (size_t)dim * T_LEN + kstart;
            bv.h[0] = (kstart      >= 0 && kstart      < T_LEN) ? *(const v8h*)pv        : z;
            bv.h[1] = (kstart + 16 >= 0 && kstart + 16 < T_LEN) ? *(const v8h*)(pv + 16) : z;
            o[s] = __builtin_amdgcn_wmma_f32_16x16x32_f16(
                false, ap.v, false, bv.v, (short)0, o[s], false, false);
        }
    }

    // --- store O tile as f16 [T][512] for the final projection GEMM ---
#pragma unroll
    for (int s = 0; s < 4; ++s)
#pragma unroll
        for (int rr = 0; rr < 8; ++rr)
            oh[(size_t)(t0 + moff + rr) * C_DIM + (h << 6) + (s << 4) + r] =
                (_Float16)o[s][rr];
}

// ---------------------------------------------------------------------------
// Host launcher
// ---------------------------------------------------------------------------
extern "C" void kernel_launch(void* const* d_in, const int* in_sizes, int n_in,
                              void* d_out, int out_size, void* d_ws, size_t ws_size,
                              hipStream_t stream) {
    (void)in_sizes; (void)n_in; (void)out_size; (void)ws_size;

    const float* x    = (const float*)d_in[0];
    const float* Wq   = (const float*)d_in[1];
    const float* Wk   = (const float*)d_in[2];
    const float* Wv   = (const float*)d_in[3];
    const float* Wo   = (const float*)d_in[4];
    const float* qw   = (const float*)d_in[5];
    const float* kw   = (const float*)d_in[6];
    const float* cosb = (const float*)d_in[7];
    const float* sinb = (const float*)d_in[8];

    char* ws = (char*)d_ws;
    // workspace layout (17.5 MiB total)
    _Float16* xh  = (_Float16*)(ws + 0);              // 2048*512 f16   = 2 MiB
    _Float16* WT  = (_Float16*)(ws + 2097152);        // 1024*512 f16   = 1 MiB
    _Float16* WoT = (_Float16*)(ws + 3145728);        //  512*512 f16   = 0.5 MiB
    float*    qkv = (float*)   (ws + 3670016);        // 2048*1024 f32  = 8 MiB
    _Float16* qh  = (_Float16*)(ws + 12058624);       // 8*2048*64 f16  = 2 MiB
    _Float16* kh  = (_Float16*)(ws + 14155776);       // 4*2048*64 f16  = 1 MiB
    _Float16* vT  = (_Float16*)(ws + 15204352);       // 4*64*2048 f16  = 1 MiB
    _Float16* oh  = (_Float16*)(ws + 16252928);       // 2048*512 f16   = 2 MiB

    // 1) f32 -> f16 conversion + weight transposition
    swa_convert_kernel<<<1024, 256, 0, stream>>>(x, Wq, Wk, Wv, Wo, xh, WT, WoT);

    // 2) QKV projection: [2048 x 512] * [512 x 1024] -> qkv f32
    //    waves: (2048/32)*(1024/64) = 1024 -> 128 blocks of 8 waves
    swa_wmma_gemm_f16<<<128, 256, 0, stream>>>(xh, WT, qkv,
                                               2048, 1024, 512, 512, 512, 1024);

    // 3) RoPE + RMS + V transpose: 2048*16 waves -> 4096 blocks of 8 waves
    swa_rope_rms_kernel<<<4096, 256, 0, stream>>>(qkv, cosb, sinb, qw, kw,
                                                  qh, kh, vT);

    // 4) attention: 8 heads * 128 query tiles = 1024 waves, 2 waves/block
    swa_attn_kernel<<<512, ATTN_WAVES * 32, 0, stream>>>(qh, kh, vT, oh);

    // 5) output projection: [2048 x 512] * [512 x 512] -> d_out f32
    //    waves: (2048/32)*(512/64) = 512 -> 64 blocks of 8 waves
    swa_wmma_gemm_f16<<<64, 256, 0, stream>>>(oh, WoT, (float*)d_out,
                                              2048, 512, 512, 512, 512, 512);
}
